// Decoder_80418967650316
// MI455X (gfx1250) — compile-verified, matrix operand order checked by reading, not run on hardware
//
#include <hip/hip_runtime.h>
#include <hip/hip_bf16.h>
#include <math.h>

typedef __attribute__((ext_vector_type(16))) _Float16 v16h;
typedef __attribute__((ext_vector_type(8)))  float    v8f;

#define DEV static __device__ __forceinline__

constexpr int cB = 32, cT = 64, cH = 512, cE = 256, cM = 512, cV = 32000;
constexpr int cF  = 2 * cH;            // 1024 encoder feature dim
constexpr int cKA = cH + cF;           // 1536 attention concat dim
constexpr int cG  = 3 * cH;            // 1536 GRU gate dim
constexpr int cU  = cH + cE + cF;      // 1792 concat(s,y,c) dim

// ---------------------------------------------------------------------------
// WMMA 16-bit fragment layout (ISA 7.12.2): lane holds row (A) / column (B);
// half-elements map to K:
//   lanes 0-15 : VGPR0-3 -> K 0..7,  VGPR4-7 -> K 16..23
//   lanes16-31 : VGPR0-3 -> K 8..15, VGPR4-7 -> K 24..31
// Gathered as 8 dword (2x b128) reads per fragment.
// ---------------------------------------------------------------------------
DEV v16h load_frag_lds(const _Float16* rowbase, int k0) {
  const int lane = threadIdx.x & 31;
  const _Float16* base = rowbase + k0 + ((lane & 16) ? 8 : 0);
  const unsigned int* b32 = reinterpret_cast<const unsigned int*>(base);
  v16h f;
  unsigned int* fw = reinterpret_cast<unsigned int*>(&f);
#pragma unroll
  for (int p = 0; p < 8; ++p)
    fw[p] = b32[((p & 4) << 1) + (p & 3)];   // dwords 0..3 then 8..11
  return f;
}

DEV void async_b128_to_lds(const _Float16* g, const _Float16* lds_ptr) {
  unsigned int l = (unsigned int)(uintptr_t)lds_ptr;   // low 32b = LDS offset
  asm volatile("global_load_async_to_lds_b128 %0, %1, off"
               :: "v"(l), "v"(g) : "memory");
}

// ---------------------------------------------------------------------------
// C[32,N] = A[32,K](f16,lda) @ Bw[N,K](f16,ldb)^T + bias
// block = 128 threads (4 waves); wave w owns 16 output columns, both 16-row
// M tiles. grid.x = N/64. K % 64 == 0.
// A/B staged in LDS via GLOBAL_LOAD_ASYNC_TO_LDS_B128, double buffered,
// exactly 6 async issues per thread per 64-K chunk -> s_wait_asynccnt 0x6
// drains the previous chunk while the next streams in.
// ---------------------------------------------------------------------------
__global__ void gemm32_wmma_lds(const _Float16* __restrict__ A, int lda,
                                const _Float16* __restrict__ Bw, int ldb,
                                const float* __restrict__ bias,
                                float* __restrict__ Cf, int ldc,
                                _Float16* __restrict__ Ch, int ldch,
                                int N, int K) {
  constexpr int CK   = 64;       // K-chunk in halves
  constexpr int LSTR = CK + 8;   // padded LDS row stride (72 halves = 144B)
  __shared__ _Float16 Alds[2][32 * LSTR];
  __shared__ _Float16 Blds[2][64 * LSTR];

  const int tid  = threadIdx.x;
  const int wave = tid >> 5;
  const int lane = tid & 31;
  const int nblk = blockIdx.x * 64;
  const int ncol = nblk + wave * 16 + (lane & 15);
  const int nchunks = K / CK;

  auto issue_chunk = [&](int c, int buf) {
    // A: 32 rows x 8 b128 segs = 256 issues -> 2 per thread
#pragma unroll
    for (int i = 0; i < 2; ++i) {
      int idx = tid * 2 + i;
      int row = idx >> 3, seg = idx & 7;
      async_b128_to_lds(A + (size_t)row * lda + c * CK + seg * 8,
                        &Alds[buf][row * LSTR + seg * 8]);
    }
    // B: 64 rows x 8 b128 segs = 512 issues -> 4 per thread
#pragma unroll
    for (int i = 0; i < 4; ++i) {
      int idx = tid * 4 + i;
      int row = idx >> 3, seg = idx & 7;
      async_b128_to_lds(Bw + (size_t)(nblk + row) * ldb + c * CK + seg * 8,
                        &Blds[buf][row * LSTR + seg * 8]);
    }
  };

  v8f acc0 = {}; v8f acc1 = {};
  issue_chunk(0, 0);

  for (int c = 0; c < nchunks; ++c) {
    const int buf = c & 1;
    if (c + 1 < nchunks) {
      issue_chunk(c + 1, buf ^ 1);
      asm volatile("s_wait_asynccnt 0x6" ::: "memory");  // chunk c landed
    } else {
      asm volatile("s_wait_asynccnt 0x0" ::: "memory");
    }
    __syncthreads();

#pragma unroll
    for (int ks = 0; ks < CK; ks += 32) {
      v16h bf = load_frag_lds(&Blds[buf][(wave * 16 + (lane & 15)) * LSTR], ks);
      v16h a0 = load_frag_lds(&Alds[buf][(lane & 15) * LSTR], ks);
      v16h a1 = load_frag_lds(&Alds[buf][(16 + (lane & 15)) * LSTR], ks);
      acc0 = __builtin_amdgcn_wmma_f32_16x16x32_f16(false, a0, false, bf,
                                                    (short)0, acc0, false, false);
      acc1 = __builtin_amdgcn_wmma_f32_16x16x32_f16(false, a1, false, bf,
                                                    (short)0, acc1, false, false);
    }
    __syncthreads();   // buffer may be overwritten next iteration's prefetch
  }

  const float bv = bias ? bias[ncol] : 0.0f;
#pragma unroll
  for (int j = 0; j < 8; ++j) {
    const int m = j + ((lane & 16) ? 8 : 0);
    const int n = ncol;
    const float v0 = acc0[j] + bv;
    const float v1 = acc1[j] + bv;
    if (Cf) {
      Cf[(size_t)m * ldc + n]        = v0;
      Cf[(size_t)(m + 16) * ldc + n] = v1;
    }
    if (Ch) {
      Ch[(size_t)m * ldch + n]        = (_Float16)v0;
      Ch[(size_t)(m + 16) * ldch + n] = (_Float16)v1;
    }
  }
}

// --------------------------- fp32 -> fp16 convert --------------------------
__global__ void cvt_f32_f16_k(const float* __restrict__ src,
                              _Float16* __restrict__ dst, int n) {
  int i = blockIdx.x * blockDim.x + threadIdx.x;
  if (i < n) dst[i] = (_Float16)src[i];
}

// ------------------ init: s = 0 (f32) and u[:,0:H] = 0 (f16) ---------------
__global__ void init_state_k(float* __restrict__ s, _Float16* __restrict__ u) {
  int i = blockIdx.x * blockDim.x + threadIdx.x;
  if (i < cB * cH) {
    s[i] = 0.0f;
    int b = i / cH, j = i % cH;
    u[(size_t)b * cU + j] = (_Float16)0.0f;
  }
}

// ---------------------------------------------------------------------------
// Attention step: scores over T, softmax, context -> u[:, H+E:] as f16.
// one block per batch row, 256 threads.
// ---------------------------------------------------------------------------
__global__ void attn_k(const float* __restrict__ h,      // [B,T,F]
                       const float* __restrict__ W_a,    // [T, H+F]
                       const float* __restrict__ b_a,    // [T]
                       const float* __restrict__ s,      // [B,H]
                       _Float16* __restrict__ u,         // [B,cU]
                       int t) {
  __shared__ float red[256];
  __shared__ float p[cT];
  const int b   = blockIdx.x;
  const int tid = threadIdx.x;
  const int tt  = tid & 63;
  const int q   = tid >> 6;                  // 4 K-quarters of 384
  const float* wrow = W_a + (size_t)tt * cKA;
  const float* hb   = h + ((size_t)b * cT + t) * cF;
  const float* sb   = s + (size_t)b * cH;

  float acc = 0.0f;
  for (int k = q * 384; k < (q + 1) * 384; ++k) {
    float x = (k < cH) ? sb[k] : hb[k - cH];
    acc += wrow[k] * x;
  }
  red[tid] = acc;
  __syncthreads();
  if (tid < cT)
    p[tt] = red[tt] + red[tt + 64] + red[tt + 128] + red[tt + 192] + b_a[tt];
  __syncthreads();

  float mx = -1e30f;
  for (int i = 0; i < cT; ++i) mx = fmaxf(mx, p[i]);
  float sm = 0.0f;
  for (int i = 0; i < cT; ++i) sm += expf(p[i] - mx);
  __syncthreads();
  if (tid < cT) p[tt] = expf(p[tt] - mx) / sm;
  __syncthreads();

  for (int f = tid; f < cF; f += 256) {
    float cv = 0.0f;
    for (int k = 0; k < cT; ++k)
      cv += p[k] * h[((size_t)b * cT + k) * cF + f];
    u[(size_t)b * cU + (cH + cE) + f] = (_Float16)cv;
  }
}

// --------------------- GRU elementwise (r,z,n gate layout) -----------------
__global__ void gru_elt_k(const float* __restrict__ gi,  // [B,3H] (bias in)
                          const float* __restrict__ gh,  // [B,3H] (bias in)
                          float* __restrict__ s,         // [B,H] in/out
                          _Float16* __restrict__ u) {    // u[:,0:H] <- s_new
  int i = blockIdx.x * blockDim.x + threadIdx.x;
  if (i >= cB * cH) return;
  int b = i / cH, j = i % cH;
  const float* gib = gi + (size_t)b * cG;
  const float* ghb = gh + (size_t)b * cG;
  float r = 1.0f / (1.0f + expf(-(gib[j] + ghb[j])));
  float z = 1.0f / (1.0f + expf(-(gib[cH + j] + ghb[cH + j])));
  float n = tanhf(gib[2 * cH + j] + r * ghb[2 * cH + j]);
  float sv = s[i];
  float sn = (1.0f - z) * n + z * sv;
  s[i] = sn;
  u[(size_t)b * cU + j] = (_Float16)sn;
}

// ------------------------------- maxout ------------------------------------
__global__ void maxout_k(const float* __restrict__ tt,   // [B,2M]
                         _Float16* __restrict__ t16) {   // [B,M]
  int i = blockIdx.x * blockDim.x + threadIdx.x;
  if (i >= cB * cM) return;
  int b = i / cM, j = i % cM;
  float v = fmaxf(tt[(size_t)b * 2 * cM + j], tt[(size_t)b * 2 * cM + cM + j]);
  t16[i] = (_Float16)v;
}

// --------------------- log-softmax row stats (max, log-sum) ----------------
__global__ void lsm_reduce_k(const float* __restrict__ logits,  // [B,V]
                             float* __restrict__ stats) {       // [B,2]
  __shared__ float red[256];
  const int b = blockIdx.x, tid = threadIdx.x;
  const float* row = logits + (size_t)b * cV;
  float mx = -1e30f;
  for (int v = tid; v < cV; v += 256) mx = fmaxf(mx, row[v]);
  red[tid] = mx; __syncthreads();
  for (int st = 128; st > 0; st >>= 1) {
    if (tid < st) red[tid] = fmaxf(red[tid], red[tid + st]);
    __syncthreads();
  }
  mx = red[0]; __syncthreads();
  float sm = 0.0f;
  for (int v = tid; v < cV; v += 256) sm += expf(row[v] - mx);
  red[tid] = sm; __syncthreads();
  for (int st = 128; st > 0; st >>= 1) {
    if (tid < st) red[tid] += red[tid + st];
    __syncthreads();
  }
  if (tid == 0) { stats[b * 2] = mx; stats[b * 2 + 1] = logf(red[0]); }
}

// ----------------------- write out[b,t,:] = logp ---------------------------
__global__ void writeout_k(const float* __restrict__ logits,
                           const float* __restrict__ stats,
                           float* __restrict__ out, int t) {
  int i = blockIdx.x * blockDim.x + threadIdx.x;
  if (i >= cB * cV) return;
  int b = i / cV, v = i % cV;
  out[((size_t)b * cT + t) * cV + v] =
      logits[i] - stats[b * 2] - stats[b * 2 + 1];
}

// ===========================================================================
extern "C" void kernel_launch(void* const* d_in, const int* in_sizes, int n_in,
                              void* d_out, int out_size, void* d_ws, size_t ws_size,
                              hipStream_t stream) {
  const float* h    = (const float*)d_in[0];   // [B,T,F]
  const float* W_a  = (const float*)d_in[1];   // [T,H+F]
  const float* b_a  = (const float*)d_in[2];   // [T]
  const float* W_ih = (const float*)d_in[3];   // [3H,F]
  const float* W_hh = (const float*)d_in[4];   // [3H,H]
  const float* b_ih = (const float*)d_in[5];   // [3H]
  const float* b_hh = (const float*)d_in[6];   // [3H]
  const float* W_emb= (const float*)d_in[7];   // [E,H]
  const float* b_emb= (const float*)d_in[8];   // [E]
  const float* W_t  = (const float*)d_in[9];   // [2M, E+3H]
  const float* b_t  = (const float*)d_in[10];  // [2M]
  const float* W_o  = (const float*)d_in[11];  // [V,M]
  const float* b_o  = (const float*)d_in[12];  // [V]
  float* out = (float*)d_out;                  // [B,T,V]

  char* ws = (char*)d_ws;
  size_t off = 0;
  auto carve = [&](size_t bytes) {
    void* p = ws + off;
    off += (bytes + 255) & ~(size_t)255;
    return p;
  };
  _Float16* Wih16  = (_Float16*)carve((size_t)cG * cF * 2);
  _Float16* Whh16  = (_Float16*)carve((size_t)cG * cH * 2);
  _Float16* Wemb16 = (_Float16*)carve((size_t)cE * cH * 2);
  _Float16* Wt16   = (_Float16*)carve((size_t)(2 * cM) * cU * 2);
  _Float16* Wo16   = (_Float16*)carve((size_t)cV * cM * 2);
  float*    sbuf   = (float*)   carve((size_t)cB * cH * 4);
  _Float16* u      = (_Float16*)carve((size_t)cB * cU * 2);
  float*    gi     = (float*)   carve((size_t)cB * cG * 4);
  float*    gh     = (float*)   carve((size_t)cB * cG * 4);
  float*    ttld   = (float*)   carve((size_t)cB * 2 * cM * 4);
  _Float16* t16    = (_Float16*)carve((size_t)cB * cM * 2);
  float*    logits = (float*)   carve((size_t)cB * cV * 4);
  float*    stats  = (float*)   carve((size_t)cB * 2 * 4);
  (void)ws_size; (void)in_sizes; (void)n_in; (void)out_size;

  auto cvt = [&](const float* src, _Float16* dst, int n) {
    cvt_f32_f16_k<<<(n + 255) / 256, 256, 0, stream>>>(src, dst, n);
  };
  cvt(W_ih,  Wih16,  cG * cF);
  cvt(W_hh,  Whh16,  cG * cH);
  cvt(W_emb, Wemb16, cE * cH);
  cvt(W_t,   Wt16,   2 * cM * cU);
  cvt(W_o,   Wo16,   cV * cM);
  init_state_k<<<(cB * cH + 255) / 256, 256, 0, stream>>>(sbuf, u);

  for (int t = 0; t < cT; ++t) {
    attn_k<<<cB, 256, 0, stream>>>(h, W_a, b_a, sbuf, u, t);

    // gi = c @ W_ih^T + b_ih
    gemm32_wmma_lds<<<cG / 64, 128, 0, stream>>>(
        u + (cH + cE), cU, Wih16, cF, b_ih, gi, cG, nullptr, 0, cG, cF);
    // gh = s_old @ W_hh^T + b_hh
    gemm32_wmma_lds<<<cG / 64, 128, 0, stream>>>(
        u, cU, Whh16, cH, b_hh, gh, cG, nullptr, 0, cG, cH);

    gru_elt_k<<<(cB * cH + 255) / 256, 256, 0, stream>>>(gi, gh, sbuf, u);

    // y = s_new @ W_emb^T + b_emb -> u[:, H:H+E] (f16)
    gemm32_wmma_lds<<<cE / 64, 128, 0, stream>>>(
        u, cU, Wemb16, cH, b_emb, nullptr, 0, u + cH, cU, cE, cH);

    // t_tilde = concat(s,y,c) @ W_t^T + b_t
    gemm32_wmma_lds<<<(2 * cM) / 64, 128, 0, stream>>>(
        u, cU, Wt16, cU, b_t, ttld, 2 * cM, nullptr, 0, 2 * cM, cU);

    maxout_k<<<(cB * cM + 255) / 256, 256, 0, stream>>>(ttld, t16);

    // logits = t @ W_o^T + b_o   (dominant GEMM, N = 32000)
    gemm32_wmma_lds<<<cV / 64, 128, 0, stream>>>(
        t16, cM, Wo16, cM, b_o, logits, cV, nullptr, 0, cV, cM);

    lsm_reduce_k<<<cB, 256, 0, stream>>>(logits, stats);
    writeout_k<<<(cB * cV + 255) / 256, 256, 0, stream>>>(logits, stats, out, t);
  }
}
